// _Int8DynActIntxWeightQuantizedLinearFallback_1812476199312
// MI455X (gfx1250) — 8-bit matrix operand order repaired and checked by reading, compiled, not run on hardware
//
#include <hip/hip_runtime.h>

typedef int   v2i __attribute__((ext_vector_type(2)));
typedef int   v4i __attribute__((ext_vector_type(4)));
typedef int   v8i __attribute__((ext_vector_type(8)));
typedef float v8f __attribute__((ext_vector_type(8)));
typedef unsigned int v4u __attribute__((ext_vector_type(4)));

#define K_DIM 4096
#define N_DIM 4096
#define GROUP 128
#define NG    (K_DIM / GROUP)

#define BM   128
#define BN   64
#define LSTR 144
#define ABUF (BM * LSTR)
#define BBUF (BN * LSTR)

#if __has_builtin(__builtin_amdgcn_tensor_load_to_lds) && \
    __has_builtin(__builtin_amdgcn_s_wait_tensorcnt)
#define USE_TDM 1
#else
#define USE_TDM 0
#endif

__global__ __launch_bounds__(256) void quant_act_kernel(
    const float* __restrict__ x, char* __restrict__ aq,
    float* __restrict__ a_scale, float* __restrict__ a_zero,
    float* __restrict__ a_gsum)
{
  __shared__ __align__(16) float row[K_DIM];
  __shared__ float rmin[256], rmax[256];
  __shared__ float gs[NG];
  __shared__ float s_scale, s_zero;

  const int r   = blockIdx.x;
  const int tid = threadIdx.x;
  const float* xr = x + (size_t)r * K_DIM;

  for (int c = tid; c < K_DIM / 4; c += 256)
    ((float4*)row)[c] = ((const float4*)xr)[c];
  __syncthreads();

  float lmin = 3.4e38f, lmax = -3.4e38f;
  for (int k = tid; k < K_DIM; k += 256) {
    float v = row[k];
    lmin = fminf(lmin, v);
    lmax = fmaxf(lmax, v);
  }
  rmin[tid] = lmin; rmax[tid] = lmax;
  __syncthreads();
  for (int s = 128; s > 0; s >>= 1) {
    if (tid < s) {
      rmin[tid] = fminf(rmin[tid], rmin[tid + s]);
      rmax[tid] = fmaxf(rmax[tid], rmax[tid + s]);
    }
    __syncthreads();
  }
  if (tid == 0) {
    float vmin = rmin[0], vmax = rmax[0];
    float sc = (vmax - vmin) / 255.0f;
    float zp = -128.0f - rintf(vmin / sc);
    s_scale = sc; s_zero = zp;
    a_scale[r] = sc; a_zero[r] = zp;
  }
  if (tid < NG) gs[tid] = 0.0f;
  __syncthreads();

  const float inv = 1.0f / s_scale;
  const float zp  = s_zero;
  char* aqr = aq + (size_t)r * K_DIM;
  for (int c = tid; c < K_DIM / 4; c += 256) {
    int packed = 0;
    float gsum = 0.0f;
#pragma unroll
    for (int e = 0; e < 4; ++e) {
      float q = rintf(row[c * 4 + e] * inv) + zp;
      q = fminf(fmaxf(q, -128.0f), 127.0f);
      int qi = (int)q;
      gsum += q;
      packed |= (qi & 0xff) << (8 * e);
    }
    ((int*)aqr)[c] = packed;
    atomicAdd(&gs[c >> 5], gsum);
  }
  __syncthreads();
  if (tid < NG) a_gsum[(size_t)r * NG + tid] = gs[tid];
}

__global__ __launch_bounds__(256) void pack_w_kernel(
    const int* __restrict__ wq32,
    const float* __restrict__ w_scales, const float* __restrict__ w_zeros,
    char* __restrict__ wq8, float* __restrict__ wP)
{
  __shared__ float gs[NG];
  __shared__ float ps[NG];
  const int n   = blockIdx.x;
  const int tid = threadIdx.x;
  if (tid < NG) gs[tid] = 0.0f;
  __syncthreads();

  const int* wr = wq32 + (size_t)n * K_DIM;
  char* wo = wq8 + (size_t)n * K_DIM;
  for (int c = tid; c < K_DIM / 4; c += 256) {
    int4 v = ((const int4*)wr)[c];
    int packed = (v.x & 0xff) | ((v.y & 0xff) << 8) |
                 ((v.z & 0xff) << 16) | (v.w << 24);
    ((int*)wo)[c] = packed;
    atomicAdd(&gs[c >> 5], (float)(v.x + v.y + v.z + v.w));
  }
  __syncthreads();
  if (tid < NG) {
    float s = w_scales[(size_t)n * NG + tid];
    float z = w_zeros[(size_t)n * NG + tid];
    ps[tid] = s * fmaf((float)GROUP, z, -gs[tid]);
  }
  __syncthreads();
  if (tid == 0) {
    float p = 0.0f;
#pragma unroll
    for (int g = 0; g < NG; ++g) p += ps[g];
    wP[n] = p;
  }
}

#if USE_TDM
__device__ __forceinline__ void tdm_load_tile(unsigned lds_off, const void* gptr,
                                              unsigned tile_rows, unsigned tensor_rows)
{
  unsigned long long ga = (unsigned long long)(uintptr_t)gptr;
  v4u g0;
  g0[0] = 1u;
  g0[1] = lds_off;
  g0[2] = (unsigned)ga;
  g0[3] = (unsigned)((ga >> 32) & 0x1FFFFFFull) | (2u << 30);
  v8i g1;
  g1[0] = (int)((1u << 20) | (4u << 22) | (3u << 25));
  g1[1] = (int)(((unsigned)K_DIM & 0xFFFFu) << 16);
  g1[2] = (int)(((unsigned)K_DIM >> 16) |
                ((tensor_rows & 0xFFFFu) << 16));
  g1[3] = (int)((tensor_rows >> 16) |
                ((unsigned)GROUP << 16));
  g1[4] = (int)(tile_rows & 0xFFFFu);
  g1[5] = (int)K_DIM;
  g1[6] = 0;
  g1[7] = 0;
  v4i z4 = {};
#if defined(__clang_major__) && (__clang_major__ >= 23)
  v8i z8 = {};
  __builtin_amdgcn_tensor_load_to_lds(g0, g1, z4, z4, z8, 0);
#else
  __builtin_amdgcn_tensor_load_to_lds(g0, g1, z4, z4, 0);
#endif
}
#endif

__global__ __launch_bounds__(256) void gemm_iu8_kernel(
    const char* __restrict__ aq, const char* __restrict__ wq8,
    const float* __restrict__ a_scale, const float* __restrict__ a_zero,
    const float* __restrict__ a_gsum,
    const float* __restrict__ w_scales, const float* __restrict__ w_zeros,
    const float* __restrict__ wP,
    float* __restrict__ out, int R)
{
  __shared__ __align__(16) char As[2 * ABUF];
  __shared__ __align__(16) char Bs[2 * BBUF];
  __shared__ float s_az[BM], s_asc[BM], s_ag[BM];
  __shared__ float s_ws[BN], s_wsz[BN];

  const int tid  = threadIdx.x;
  const int lane = tid & 31;
  const int wave = tid >> 5;
  const int wrow = wave >> 1;
  const int wcol = wave & 1;
  const int rb = blockIdx.y * BM;
  const int nb = blockIdx.x * BN;
  const int h  = lane >> 4;
  const int ml = lane & 15;

  if (tid < BM) {
    s_az[tid]  = a_zero[rb + tid];
    s_asc[tid] = a_scale[rb + tid];
  }

#if USE_TDM
  const unsigned ldsA = (unsigned)(uintptr_t)(void*)&As[0];
  const unsigned ldsB = (unsigned)(uintptr_t)(void*)&Bs[0];
  if (wave == 0) {
    tdm_load_tile(ldsA, aq  + (size_t)rb * K_DIM, BM, (unsigned)R);
    tdm_load_tile(ldsB, wq8 + (size_t)nb * K_DIM, BN, N_DIM);
  }
#endif

  v8f F[2][2] = {};
  v8f T[2][2] = {};

  for (int g = 0; g < NG; ++g) {
    const int cur = g & 1;

    if (tid < BM) s_ag[tid] = a_gsum[(size_t)(rb + tid) * NG + g];
    const int t2 = tid - BM;
    if (t2 >= 0 && t2 < BN) {
      float s = w_scales[(size_t)(nb + t2) * NG + g];
      float z = w_zeros[(size_t)(nb + t2) * NG + g];
      s_ws[t2]  = s;
      s_wsz[t2] = s * z;
    }

#if USE_TDM
    if (g + 1 < NG) {
      if (wave == 0) {
        const int kg1 = (g + 1) * GROUP;
        tdm_load_tile(ldsA + (unsigned)((1 - cur) * ABUF),
                      aq  + (size_t)rb * K_DIM + kg1, BM, (unsigned)R);
        tdm_load_tile(ldsB + (unsigned)((1 - cur) * BBUF),
                      wq8 + (size_t)nb * K_DIM + kg1, BN, N_DIM);
      }
      __builtin_amdgcn_s_wait_tensorcnt(2);
    } else {
      __builtin_amdgcn_s_wait_tensorcnt(0);
    }
#else
    {
      const int kg = g * GROUP;
      for (int c = tid; c < BM * 8; c += 256) {
        int row = c >> 3, ch = c & 7;
        *(int4*)(As + cur * ABUF + row * LSTR + ch * 16) =
            *(const int4*)(aq + (size_t)(rb + row) * K_DIM + kg + ch * 16);
      }
      for (int c = tid; c < BN * 8; c += 256) {
        int row = c >> 3, ch = c & 7;
        *(int4*)(Bs + cur * BBUF + row * LSTR + ch * 16) =
            *(const int4*)(wq8 + (size_t)(nb + row) * K_DIM + kg + ch * 16);
      }
    }
#endif
    __syncthreads();

    const char* Ab = As + cur * ABUF;
    const char* Bb = Bs + cur * BBUF;

    v8i C[2][2] = {};
#pragma unroll
    for (int kk = 0; kk < GROUP; kk += 64) {
      v8i a[2], b[2];
#pragma unroll
      for (int i = 0; i < 2; ++i) {
        const char* base = Ab + (wrow * 32 + i * 16 + ml) * LSTR + kk + h * 16;
#pragma unroll
        for (int q = 0; q < 4; ++q) {
          v2i t = *(const v2i*)(base + (q & 1) * 8 + (q >> 1) * 32);
          a[i][2 * q]     = t[0];
          a[i][2 * q + 1] = t[1];
        }
      }
#pragma unroll
      for (int j = 0; j < 2; ++j) {
        const char* base = Bb + (wcol * 32 + j * 16 + ml) * LSTR + kk + h * 16;
        v4i t0 = *(const v4i*)(base);
        v4i t1 = *(const v4i*)(base + 32);
        b[j][0] = t0[0]; b[j][1] = t0[1]; b[j][2] = t0[2]; b[j][3] = t0[3];
        b[j][4] = t1[0]; b[j][5] = t1[1]; b[j][6] = t1[2]; b[j][7] = t1[3];
      }
#pragma unroll
      for (int i = 0; i < 2; ++i)
#pragma unroll
        for (int j = 0; j < 2; ++j)
          C[i][j] = __builtin_amdgcn_wmma_i32_16x16x64_iu8(
              true, a[i], true, b[j], C[i][j], false, false);
    }

#pragma unroll
    for (int j = 0; j < 2; ++j) {
      const int n_l = wcol * 32 + j * 16 + ml;
      const float s  = s_ws[n_l];
      const float sz = s_wsz[n_l];
#pragma unroll
      for (int i = 0; i < 2; ++i) {
#pragma unroll
        for (int v = 0; v < 8; ++v) {
          const int m_l = wrow * 32 + i * 16 + v + 8 * h;
          F[i][j][v] = fmaf(s, (float)C[i][j][v], F[i][j][v]);
          T[i][j][v] = fmaf(sz, s_ag[m_l], T[i][j][v]);
        }
      }
    }
    __syncthreads();
  }

#pragma unroll
  for (int j = 0; j < 2; ++j) {
    const int n_l = wcol * 32 + j * 16 + ml;
    const float Pn = wP[nb + n_l];
#pragma unroll
    for (int i = 0; i < 2; ++i) {
#pragma unroll
      for (int v = 0; v < 8; ++v) {
        const int m_l = wrow * 32 + i * 16 + v + 8 * h;
        const float r = F[i][j][v] - T[i][j][v] + s_az[m_l] * Pn;
        out[(size_t)(rb + m_l) * N_DIM + (nb + n_l)] = s_asc[m_l] * r;
      }
    }
  }
}

extern "C" void kernel_launch(void* const* d_in, const int* in_sizes, int n_in,
                              void* d_out, int out_size, void* d_ws, size_t ws_size,
                              hipStream_t stream) {
  const float* x    = (const float*)d_in[0];
  const int*   wq32 = (const int*)d_in[1];
  const float* wsc  = (const float*)d_in[2];
  const float* wz   = (const float*)d_in[3];
  const int R = in_sizes[0] / K_DIM;

  char*  aq      = (char*)d_ws;
  char*  wq8     = aq + (size_t)R * K_DIM;
  float* a_scale = (float*)(wq8 + (size_t)N_DIM * K_DIM);
  float* a_zero  = a_scale + R;
  float* a_gsum  = a_zero + R;
  float* wP      = a_gsum + (size_t)R * NG;

  quant_act_kernel<<<R, 256, 0, stream>>>(x, aq, a_scale, a_zero, a_gsum);
  pack_w_kernel<<<N_DIM, 256, 0, stream>>>(wq32, wsc, wz, wq8, wP);

  dim3 grid(N_DIM / BN, R / BM);
  gemm_iu8_kernel<<<grid, 256, 0, stream>>>(aq, wq8, a_scale, a_zero, a_gsum,
                                            wsc, wz, wP, (float*)d_out, R);
}
